// AAConv2D_86655260164817
// MI455X (gfx1250) — matrix operand order repaired and checked by reading, not compiled, not run on hardware
//
#include <hip/hip_runtime.h>
#include <hip/hip_bf16.h>

typedef __attribute__((ext_vector_type(16))) _Float16 v16h;
typedef __attribute__((ext_vector_type(8)))  float    v8f;

#define Bsz 8
#define Himg 32
#define Wimg 32
#define FIN 256
#define FOUT 512
#define DK 256
#define DV 256
#define NH 8
#define DKH 32
#define DVH 32
#define HWSZ 1024
#define CONVF (FOUT - DV)   // 256
#define KK 2304             // 3*3*256

// ---------------- fragment loaders (ISA 7.12.2 layouts, f16 16x16x32) -------
// A (16xK=32), stored row-major [M][ld], K contiguous.
// lane L: M = L&15, half = L>>4; elems 0..7 -> K=half*8+e; 8..15 -> K=16+half*8+(e-8)
__device__ __forceinline__ v16h load_a_frag(const _Float16* tile, int ld) {
  int lane = threadIdx.x & 31;
  int r = lane & 15, hh = lane >> 4;
  const _Float16* p = tile + (size_t)r * ld;
  v16h f;
  __builtin_memcpy(&f, p + hh * 8, 16);
  __builtin_memcpy(reinterpret_cast<char*>(&f) + 16, p + 16 + hh * 8, 16);
  return f;
}
// B (K=32 x 16), stored transposed row-major [N][ld], K contiguous.
// lane L: N = L&15, half = L>>4; elems 0..7 -> K=half*8+e; 8..15 -> K=16+half*8+(e-8)
__device__ __forceinline__ v16h load_b_frag(const _Float16* tile, int ld) {
  int lane = threadIdx.x & 31;
  int r = lane & 15, hh = lane >> 4;
  const _Float16* p = tile + (size_t)r * ld + hh * 8;
  v16h f;
  __builtin_memcpy(&f, p, 16);
  __builtin_memcpy(reinterpret_cast<char*>(&f) + 16, p + 16, 16);
  return f;
}
__device__ __forceinline__ v8f wmma16(v16h a, v16h b, v8f c) {
  return __builtin_amdgcn_wmma_f32_16x16x32_f16(false, a, false, b, (short)0, c,
                                                false, false);
}

// ---------------- kernel 1: f32 -> f16 conversions + weight transposes ------
__global__ __launch_bounds__(256) void aac_prep(
    const float* __restrict__ x, const float* __restrict__ wkqv,
    const float* __restrict__ wconv, const float* __restrict__ wproj,
    _Float16* __restrict__ xh, _Float16* __restrict__ wkqvh,
    _Float16* __restrict__ wch, _Float16* __restrict__ wph) {
  const int NX = Bsz * Himg * Wimg * FIN;   // 2,097,152
  const int NK = FIN * (2 * DK + DV);       // 196,608
  const int NC = KK * CONVF;                // 589,824
  const int NP = DV * DV;                   // 65,536
  for (int idx = blockIdx.x * blockDim.x + threadIdx.x; idx < NX + NK + NC + NP;
       idx += gridDim.x * blockDim.x) {
    if (idx < NX) {
      xh[idx] = (_Float16)x[idx];
    } else if (idx < NX + NK) {
      int t = idx - NX;
      int o = t / FIN, c = t % FIN;                 // wkqvh[o][c] = wkqv[c][o]
      wkqvh[t] = (_Float16)wkqv[(size_t)c * (2 * DK + DV) + o];
    } else if (idx < NX + NK + NC) {
      int t = idx - NX - NK;
      int co = t / KK, kkci = t % KK;               // wch[co][kh*3*FIN+kw*FIN+ci]
      wch[t] = (_Float16)wconv[(size_t)kkci * CONVF + co];
    } else {
      int t = idx - NX - NK - NC;
      int o = t / DV, c = t % DV;                   // wph[o][c] = wproj[c][o]
      wph[t] = (_Float16)wproj[(size_t)c * DV + o];
    }
  }
}

// ---------------- kernel 2: KQV GEMM (8192 x 768 x 256), 32x32/wave ---------
__global__ __launch_bounds__(256) void aac_kqv(
    const _Float16* __restrict__ xh, const _Float16* __restrict__ wk,
    const float* __restrict__ bias, _Float16* __restrict__ khw,
    _Float16* __restrict__ qhw, _Float16* __restrict__ vtw) {
  int wave = threadIdx.x >> 5, lane = threadIdx.x & 31;
  int hh = lane >> 4, nn = lane & 15;
  int m0 = blockIdx.x * 64 + (wave & 1) * 32;
  int n0 = blockIdx.y * 128 + (wave >> 1) * 32;
  const _Float16* A0 = xh + (size_t)m0 * FIN;
  const _Float16* B0 = wk + (size_t)n0 * FIN;
  v8f acc[2][2] = {};
  for (int k0 = 0; k0 < FIN; k0 += 32) {
    v16h a0 = load_a_frag(A0 + k0, FIN);
    v16h a1 = load_a_frag(A0 + (size_t)16 * FIN + k0, FIN);
    v16h b0 = load_b_frag(B0 + k0, FIN);
    v16h b1 = load_b_frag(B0 + (size_t)16 * FIN + k0, FIN);
    acc[0][0] = wmma16(a0, b0, acc[0][0]);
    acc[0][1] = wmma16(a0, b1, acc[0][1]);
    acc[1][0] = wmma16(a1, b0, acc[1][0]);
    acc[1][1] = wmma16(a1, b1, acc[1][1]);
  }
  for (int mt = 0; mt < 2; ++mt)
    for (int nt = 0; nt < 2; ++nt) {
      int o = n0 + nt * 16 + nn;
      float bv = bias[o];
      for (int r = 0; r < 8; ++r) {
        int gi = m0 + mt * 16 + hh * 8 + r;   // global spatial row
        int bi = gi >> 10, ip = gi & 1023;
        float val = acc[mt][nt][r] + bv;
        if (o < DK) {                         // K
          int head = o >> 5, d = o & 31;
          khw[(((size_t)bi * NH + head) * HWSZ + ip) * DKH + d] = (_Float16)val;
        } else if (o < 2 * DK) {              // Q (pre-scale by DKH^-0.5)
          int oq = o - DK, head = oq >> 5, d = oq & 31;
          qhw[(((size_t)bi * NH + head) * HWSZ + ip) * DKH + d] =
              (_Float16)(val * 0.17677669529663687f);
        } else {                              // V (transposed [d][j])
          int ov = o - 2 * DK, head = ov >> 5, d = ov & 31;
          vtw[(((size_t)bi * NH + head) * DVH + d) * HWSZ + ip] = (_Float16)val;
        }
      }
    }
}

// ---------------- kernel 3: 3x3 conv as implicit GEMM, 32x32/wave -----------
__global__ __launch_bounds__(256) void aac_conv(
    const _Float16* __restrict__ xh, const _Float16* __restrict__ wc,
    const float* __restrict__ bias, float* __restrict__ out) {
  __shared__ _Float16 xrow[34 * FIN];   // padded input row, 17 KB
  int by = blockIdx.x;                  // b*32 + y
  int b = by >> 5, y = by & 31;
  int wave = threadIdx.x >> 5, lane = threadIdx.x & 31;
  int hh = lane >> 4, nn = lane & 15;
  int n0 = wave * 32;                   // 8 waves cover all 256 co
  v8f acc[2][2] = {};
  for (int kh = 0; kh < 3; ++kh) {
    int ys = y + kh - 1;
    __syncthreads();
    for (int chunk = threadIdx.x; chunk < 34 * (FIN / 8); chunk += 256) {
      int col = chunk >> 5;             // 0..33  (x = col-1)
      int ci8 = (chunk & 31) * 8;
      int xs = col - 1;
      _Float16 tmp[8] = {};
      if (ys >= 0 && ys < Himg && xs >= 0 && xs < Wimg)
        __builtin_memcpy(tmp, xh + (((size_t)(b * Himg + ys)) * Wimg + xs) * FIN + ci8, 16);
      __builtin_memcpy(xrow + col * FIN + ci8, tmp, 16);
    }
    __syncthreads();
    for (int kw = 0; kw < 3; ++kw) {
      const _Float16* Ab = xrow + kw * FIN;   // col = x + kw - 1 (+1 pad)
      const _Float16* Bb = wc + (size_t)n0 * KK + (kh * 3 + kw) * FIN;
      for (int k0 = 0; k0 < FIN; k0 += 32) {
        v16h a0 = load_a_frag(Ab + k0, FIN);                       // x = 0..15
        v16h a1 = load_a_frag(Ab + (size_t)16 * FIN + k0, FIN);    // x = 16..31
        v16h b0 = load_b_frag(Bb + k0, KK);
        v16h b1 = load_b_frag(Bb + (size_t)16 * KK + k0, KK);
        acc[0][0] = wmma16(a0, b0, acc[0][0]);
        acc[0][1] = wmma16(a0, b1, acc[0][1]);
        acc[1][0] = wmma16(a1, b0, acc[1][0]);
        acc[1][1] = wmma16(a1, b1, acc[1][1]);
      }
    }
  }
  for (int mt = 0; mt < 2; ++mt)
    for (int nt = 0; nt < 2; ++nt) {
      int co = n0 + nt * 16 + nn;
      float bv = bias[co];
      for (int r = 0; r < 8; ++r) {
        int x = mt * 16 + hh * 8 + r;
        size_t gi = (size_t)by * Wimg + x;
        out[gi * FOUT + co] = acc[mt][nt][r] + bv;   // channels [0, 256)
      }
    }
}

// ---------------- kernel 4: fused attention (flash-style, per head) ---------
__global__ __launch_bounds__(128) void aac_attn(
    const _Float16* __restrict__ qh, const _Float16* __restrict__ kh,
    const _Float16* __restrict__ vt, const float* __restrict__ relw,
    const float* __restrict__ relh, _Float16* __restrict__ attn) {
  int bn = blockIdx.x;                  // b*NH + nh
  int blk = blockIdx.y;                 // 64-row block, 16 per head-image
  int b = bn >> 3, nh = bn & 7;
  int tid = threadIdx.x, wave = tid >> 5, lane = tid & 31;
  int hh = lane >> 4, nn = lane & 15;

  __shared__ float rw_s[64][63];
  __shared__ float rh_s[64][63];
  __shared__ _Float16 p_s[4][16][32];

  const _Float16* qbase = qh + (((size_t)bn * HWSZ) + blk * 64) * DKH;
  // exact relative-logit tables: rw_s[i][m] = q_i . key_rel_w[m]  (q pre-scaled)
  for (int e = tid; e < 64 * 63; e += 128) {
    int ii = e / 63, m = e % 63;
    const _Float16* qr = qbase + ii * DKH;
    float aw = 0.f, ah = 0.f;
    for (int d = 0; d < DKH; ++d) {
      float qv = (float)qr[d];
      aw += qv * relw[m * DKH + d];
      ah += qv * relh[m * DKH + d];
    }
    rw_s[ii][m] = aw;
    rh_s[ii][m] = ah;
  }
  __syncthreads();

  int rowLoc0 = wave * 16;              // this wave's 16 rows within the block
  v16h qa = load_a_frag(qbase + (size_t)rowLoc0 * DKH, DKH);
  const _Float16* kbase = kh + (size_t)bn * HWSZ * DKH;
  const _Float16* vbase = vt + (size_t)bn * DVH * HWSZ;

  float mrow[8], lrow[8];
  v8f o0 = {}, o1 = {};
  for (int r = 0; r < 8; ++r) { mrow[r] = -1e30f; lrow[r] = 0.f; }

  for (int jb = 0; jb < HWSZ; jb += 32) {
    v16h kf0 = load_b_frag(kbase + (size_t)jb * DKH, DKH);
    v16h kf1 = load_b_frag(kbase + (size_t)(jb + 16) * DKH, DKH);
    v8f s0 = {}, s1 = {};
    s0 = wmma16(qa, kf0, s0);           // logits cols jb..jb+15
    s1 = wmma16(qa, kf1, s1);           // logits cols jb+16..jb+31
    int j0 = jb + nn, j1 = j0 + 16;
    int xj0 = j0 >> 5, yj0 = j0 & 31;
    int xj1 = j1 >> 5, yj1 = j1 & 31;
    for (int r = 0; r < 8; ++r) {
      int iLoc = rowLoc0 + hh * 8 + r;
      int ig = blk * 64 + iLoc;
      int xi = ig >> 5, yi = ig & 31;
      s0[r] += rw_s[iLoc][yj0 - yi + 31] + rh_s[iLoc][xj0 - xi + 31];
      s1[r] += rw_s[iLoc][yj1 - yi + 31] + rh_s[iLoc][xj1 - xi + 31];
    }
    // online softmax over the 32 new logits per row (row = 16 lanes of half)
    for (int r = 0; r < 8; ++r) {
      float v = fmaxf(s0[r], s1[r]);
      for (int off = 1; off < 16; off <<= 1) v = fmaxf(v, __shfl_xor(v, off, 32));
      float mnew = fmaxf(mrow[r], v);
      float p0 = __expf(s0[r] - mnew), p1 = __expf(s1[r] - mnew);
      float ss = p0 + p1;
      for (int off = 1; off < 16; off <<= 1) ss += __shfl_xor(ss, off, 32);
      float sc = __expf(mrow[r] - mnew);
      lrow[r] = lrow[r] * sc + ss;
      mrow[r] = mnew;
      o0[r] *= sc; o1[r] *= sc;
      int M = hh * 8 + r;
      p_s[wave][M][nn] = (_Float16)p0;
      p_s[wave][M][16 + nn] = (_Float16)p1;
    }
    v16h pa = load_a_frag(&p_s[wave][0][0], 32);
    v16h vf0 = load_b_frag(vbase + jb, HWSZ);                     // d = 0..15
    v16h vf1 = load_b_frag(vbase + (size_t)16 * HWSZ + jb, HWSZ); // d = 16..31
    o0 = wmma16(pa, vf0, o0);
    o1 = wmma16(pa, vf1, o1);
  }
  for (int r = 0; r < 8; ++r) {
    int ig = blk * 64 + rowLoc0 + hh * 8 + r;
    size_t row = (size_t)b * HWSZ + ig;
    float inv = 1.f / lrow[r];
    attn[row * DV + nh * DVH + nn] = (_Float16)(o0[r] * inv);
    attn[row * DV + nh * DVH + 16 + nn] = (_Float16)(o1[r] * inv);
  }
}

// ---------------- kernel 5: output projection (8192 x 256 x 256) ------------
__global__ __launch_bounds__(256) void aac_proj(
    const _Float16* __restrict__ attn, const _Float16* __restrict__ wp,
    const float* __restrict__ bias, float* __restrict__ out) {
  int wave = threadIdx.x >> 5, lane = threadIdx.x & 31;
  int hh = lane >> 4, nn = lane & 15;
  int m0 = blockIdx.x * 64 + (wave & 1) * 32;
  int n0 = blockIdx.y * 128 + (wave >> 1) * 32;
  const _Float16* A0 = attn + (size_t)m0 * DV;
  const _Float16* B0 = wp + (size_t)n0 * DV;
  v8f acc[2][2] = {};
  for (int k0 = 0; k0 < DV; k0 += 32) {
    v16h a0 = load_a_frag(A0 + k0, DV);
    v16h a1 = load_a_frag(A0 + (size_t)16 * DV + k0, DV);
    v16h b0 = load_b_frag(B0 + k0, DV);
    v16h b1 = load_b_frag(B0 + (size_t)16 * DV + k0, DV);
    acc[0][0] = wmma16(a0, b0, acc[0][0]);
    acc[0][1] = wmma16(a0, b1, acc[0][1]);
    acc[1][0] = wmma16(a1, b0, acc[1][0]);
    acc[1][1] = wmma16(a1, b1, acc[1][1]);
  }
  for (int mt = 0; mt < 2; ++mt)
    for (int nt = 0; nt < 2; ++nt) {
      int o = n0 + nt * 16 + nn;
      float bv = bias[o];
      for (int r = 0; r < 8; ++r) {
        size_t gi = (size_t)(m0 + mt * 16 + hh * 8 + r);
        out[gi * FOUT + CONVF + o] = acc[mt][nt][r] + bv;   // channels [256, 512)
      }
    }
}

// ---------------- host side -------------------------------------------------
extern "C" void kernel_launch(void* const* d_in, const int* in_sizes, int n_in,
                              void* d_out, int out_size, void* d_ws, size_t ws_size,
                              hipStream_t stream) {
  (void)in_sizes; (void)n_in; (void)out_size; (void)ws_size;
  const float* x       = (const float*)d_in[0];
  const float* w_kqv   = (const float*)d_in[1];
  const float* b_kqv   = (const float*)d_in[2];
  const float* w_proj  = (const float*)d_in[3];
  const float* b_proj  = (const float*)d_in[4];
  const float* w_conv  = (const float*)d_in[5];
  const float* b_conv  = (const float*)d_in[6];
  const float* relw    = (const float*)d_in[7];
  const float* relh    = (const float*)d_in[8];
  float* out = (float*)d_out;

  char* base = (char*)d_ws;
  size_t off = 0;
  auto carve = [&](size_t bytes) {
    char* p = base + off;
    off = (off + bytes + 255) & ~(size_t)255;
    return p;
  };
  _Float16* xh     = (_Float16*)carve((size_t)Bsz * HWSZ * FIN * 2);
  _Float16* wkqvh  = (_Float16*)carve((size_t)(2 * DK + DV) * FIN * 2);
  _Float16* wch    = (_Float16*)carve((size_t)CONVF * KK * 2);
  _Float16* wph    = (_Float16*)carve((size_t)DV * DV * 2);
  _Float16* khw    = (_Float16*)carve((size_t)Bsz * NH * HWSZ * DKH * 2);
  _Float16* qhw    = (_Float16*)carve((size_t)Bsz * NH * HWSZ * DKH * 2);
  _Float16* vtw    = (_Float16*)carve((size_t)Bsz * NH * DVH * HWSZ * 2);
  _Float16* attnw  = (_Float16*)carve((size_t)Bsz * HWSZ * DV * 2);

  aac_prep<<<2048, 256, 0, stream>>>(x, w_kqv, w_conv, w_proj, xh, wkqvh, wch, wph);
  aac_kqv<<<dim3(Bsz * HWSZ / 64, (2 * DK + DV) / 128), 256, 0, stream>>>(
      xh, wkqvh, b_kqv, khw, qhw, vtw);
  aac_conv<<<dim3(Bsz * Himg), 256, 0, stream>>>(xh, wch, b_conv, out);
  aac_attn<<<dim3(Bsz * NH, HWSZ / 64), 128, 0, stream>>>(qhw, khw, vtw, relw, relh, attnw);
  aac_proj<<<dim3(Bsz * HWSZ / 64, DV / 128), 256, 0, stream>>>(attnw, wph, b_proj, out);
}
